// GRUModel_20461224198163
// MI455X (gfx1250) — 8-bit matrix operand order repaired and checked by reading, compiled, not run on hardware
//
#include <hip/hip_runtime.h>


typedef __attribute__((ext_vector_type(16))) int      v16i;
typedef __attribute__((ext_vector_type(8)))  float    v8f;
typedef __attribute__((ext_vector_type(4)))  int      v4i;
typedef __attribute__((ext_vector_type(4)))  _Float16 v4h;
typedef __attribute__((ext_vector_type(2)))  _Float16 v2h;

#define NTPW 6
#define TSTEPS 1024
#define NWG 32

__device__ __forceinline__ unsigned f32_to_fp8_sw(float f) {
  unsigned u = __float_as_uint(f);
  unsigned s = (u >> 24) & 0x80u;
  int e = (int)((u >> 23) & 0xffu) - 127 + 7;
  unsigned m = (u >> 20) & 0x7u;
  m += (u >> 19) & 1u;
  if (m > 7u) { m = 0u; e += 1; }
  if (e <= 0) return s;
  if (e > 15) { e = 15; m = 6u; }
  return s | ((unsigned)e << 3) | m;
}

__device__ __forceinline__ unsigned char pack_fp8(float f) {
#if __has_builtin(__builtin_amdgcn_cvt_pk_fp8_f32)
  return (unsigned char)(__builtin_amdgcn_cvt_pk_fp8_f32(f, f, 0, false) & 0xff);
#else
  return (unsigned char)f32_to_fp8_sw(f);
#endif
}

__device__ __forceinline__ float sigm(float x) { return 1.f / (1.f + __expf(-x)); }
__device__ __forceinline__ float tanh_fast(float y) {
  float e = __expf(2.f * y);
  return 1.f - 2.f / (e + 1.f);
}

__global__ void pack_wb(const float* __restrict__ whh0,
                        const float* __restrict__ whh1,
                        const float* __restrict__ wih1,
                        unsigned* __restrict__ out) {
  int g = blockIdx.x * blockDim.x + threadIdx.x;
  if (g >= 3 * 48 * 2 * 512) return;
  int mat = g / 49152, r = g % 49152;
  const float* w = (mat == 0) ? whh0 : (mat == 1) ? whh1 : wih1;
  int f = r >> 9, rem = r & 511;
  int lane = rem >> 4, v = rem & 15;
  int ntile = f >> 1, kf = f & 1;
  int n = ntile * 16 + (lane & 15);
  unsigned dw = 0;
  #pragma unroll
  for (int b = 0; b < 4; ++b) {
    int K = kf * 128 + ((v >> 2) << 5) + ((lane >> 4) << 4) + ((v & 3) << 2) + b;
    dw |= (unsigned)pack_fp8(w[n * 256 + K]) << (8 * b);
  }
  out[g] = dw;
}

__device__ __forceinline__ int a_byte_base(int c) {
  int kf = c >> 7, Kc = c & 127;
  int v = ((Kc >> 5) << 2) + ((Kc >> 2) & 3);
  int khalf = (Kc >> 4) & 1;
  return kf * 2048 + khalf * 1024 + v * 4 + (Kc & 3);
}

__global__ __launch_bounds__(256) void gru_layer0(
    const float* __restrict__ x,
    const float* __restrict__ w_ih,
    const float* __restrict__ b_ih,
    const float* __restrict__ b_hh,
    const v4i*  __restrict__ whh_frag,
    v4i* __restrict__ h0_stream)
{
  __shared__ __align__(64) int afrag[1024];
  __shared__ __align__(16) _Float16 pre[16 * 256 * 4];
  __shared__ float xs[16];

  const int tid = threadIdx.x, wg = blockIdx.x;
  const int wave = tid >> 5, lane = tid & 31;

  v16i bw[NTPW][2];
  #pragma unroll
  for (int jj = 0; jj < NTPW; ++jj)
    #pragma unroll
    for (int kf = 0; kf < 2; ++kf)
      bw[jj][kf] = *(const v16i*)(whh_frag + ((wave * NTPW + jj) * 2 + kf) * 128 + lane * 4);

  for (int i = tid; i < 1024; i += 256) afrag[i] = 0;

  const int c = tid;
  const float wir = w_ih[c], wiz = w_ih[256 + c], win = w_ih[512 + c];
  const float bir = b_ih[c], biz = b_ih[256 + c], bin = b_ih[512 + c];
  const float bhr = b_hh[c], bhz = b_hh[256 + c], bhn = b_hh[512 + c];
  const int abase = a_byte_base(c);
  unsigned char* af8 = (unsigned char*)afrag;
  float h[16];
  #pragma unroll
  for (int b = 0; b < 16; ++b) h[b] = 0.f;

  __syncthreads();

  for (int t = 0; t < TSTEPS; ++t) {
    v8f acc[NTPW];
    #pragma unroll
    for (int jj = 0; jj < NTPW; ++jj)
      #pragma unroll
      for (int r = 0; r < 8; ++r) acc[jj][r] = 0.f;

    v16i a0 = *(const v16i*)&afrag[lane * 16];
    v16i a1 = *(const v16i*)&afrag[512 + lane * 16];
    #pragma unroll
    for (int jj = 0; jj < NTPW; ++jj) {
      acc[jj] = __builtin_amdgcn_wmma_f32_16x16x128_fp8_fp8(a0, bw[jj][0], (short)0, acc[jj], false, false);
      acc[jj] = __builtin_amdgcn_wmma_f32_16x16x128_fp8_fp8(a1, bw[jj][1], (short)0, acc[jj], false, false);
    }
    {
      const int nn = (lane & 15), mb = (lane < 16) ? 0 : 8;
      #pragma unroll
      for (int jj = 0; jj < NTPW; ++jj) {
        const int ncol = (wave * NTPW + jj) * 16 + nn;
        const int gate = ncol >> 8, cc = ncol & 255;
        #pragma unroll
        for (int r = 0; r < 8; ++r)
          pre[((mb + r) * 256 + cc) * 4 + gate] = (_Float16)acc[jj][r];
      }
    }
    if (tid < 16) xs[tid] = x[(wg * 16 + tid) * 1024 + t];
    __syncthreads();

    #pragma unroll
    for (int b = 0; b < 16; ++b) {
      v4h pv = *(const v4h*)&pre[(b * 256 + c) * 4];
      float xv = xs[b];
      float r = sigm(xv * wir + bir + (float)pv[0] + bhr);
      float z = sigm(xv * wiz + biz + (float)pv[1] + bhz);
      float n = tanh_fast(xv * win + bin + r * ((float)pv[2] + bhn));
      h[b] = (1.f - z) * n + z * h[b];
      af8[abase + b * 64] = pack_fp8(h[b]);
    }
    __syncthreads();

    h0_stream[(wg * TSTEPS + t) * 256 + tid] = *(const v4i*)&afrag[tid * 4];
  }
}

__global__ __launch_bounds__(256) void gru_layer1(
    const float* __restrict__ b_ih,
    const float* __restrict__ b_hh,
    const v4i*  __restrict__ whh_frag,
    const v4i*  __restrict__ wih_frag,
    const v4i*  __restrict__ h0_stream,
    float* __restrict__ h_last)
{
  __shared__ __align__(64) int afrag[1024];
  __shared__ __align__(16) _Float16 pre[16 * 256 * 6];

  const int tid = threadIdx.x, wg = blockIdx.x;
  const int wave = tid >> 5, lane = tid & 31;

  v16i bw[NTPW][2];
  #pragma unroll
  for (int jj = 0; jj < NTPW; ++jj)
    #pragma unroll
    for (int kf = 0; kf < 2; ++kf)
      bw[jj][kf] = *(const v16i*)(whh_frag + ((wave * NTPW + jj) * 2 + kf) * 128 + lane * 4);

  for (int i = tid; i < 1024; i += 256) afrag[i] = 0;

  const int c = tid;
  const float bir = b_ih[c], biz = b_ih[256 + c], bin = b_ih[512 + c];
  const float bhr = b_hh[c], bhz = b_hh[256 + c], bhn = b_hh[512 + c];
  const int abase = a_byte_base(c);
  unsigned char* af8 = (unsigned char*)afrag;
  float h[16];
  #pragma unroll
  for (int b = 0; b < 16; ++b) h[b] = 0.f;

  __syncthreads();

  for (int t = 0; t < TSTEPS; ++t) {
    const v4i* blk = h0_stream + (wg * TSTEPS + t) * 256;
    v16i ax0 = *(const v16i*)(blk + lane * 4);
    v16i ax1 = *(const v16i*)(blk + 128 + lane * 4);
    if (t + 1 < TSTEPS)
      __builtin_prefetch((const void*)(blk + 256 + lane * 4), 0, 1);

    v16i a0 = *(const v16i*)&afrag[lane * 16];
    v16i a1 = *(const v16i*)&afrag[512 + lane * 16];

    v8f accH[NTPW], accX[NTPW];
    #pragma unroll
    for (int jj = 0; jj < NTPW; ++jj)
      #pragma unroll
      for (int r = 0; r < 8; ++r) { accH[jj][r] = 0.f; accX[jj][r] = 0.f; }

    #pragma unroll
    for (int jj = 0; jj < NTPW; ++jj) {
      accH[jj] = __builtin_amdgcn_wmma_f32_16x16x128_fp8_fp8(a0, bw[jj][0], (short)0, accH[jj], false, false);
      accH[jj] = __builtin_amdgcn_wmma_f32_16x16x128_fp8_fp8(a1, bw[jj][1], (short)0, accH[jj], false, false);
      v16i bx0 = *(const v16i*)(wih_frag + ((wave * NTPW + jj) * 2 + 0) * 128 + lane * 4);
      v16i bx1 = *(const v16i*)(wih_frag + ((wave * NTPW + jj) * 2 + 1) * 128 + lane * 4);
      accX[jj] = __builtin_amdgcn_wmma_f32_16x16x128_fp8_fp8(ax0, bx0, (short)0, accX[jj], false, false);
      accX[jj] = __builtin_amdgcn_wmma_f32_16x16x128_fp8_fp8(ax1, bx1, (short)0, accX[jj], false, false);
    }
    {
      const int nn = (lane & 15), mb = (lane < 16) ? 0 : 8;
      #pragma unroll
      for (int jj = 0; jj < NTPW; ++jj) {
        const int ncol = (wave * NTPW + jj) * 16 + nn;
        const int gate = ncol >> 8, cc = ncol & 255;
        #pragma unroll
        for (int r = 0; r < 8; ++r) {
          pre[((mb + r) * 256 + cc) * 6 + gate * 2    ] = (_Float16)accH[jj][r];
          pre[((mb + r) * 256 + cc) * 6 + gate * 2 + 1] = (_Float16)accX[jj][r];
        }
      }
    }
    __syncthreads();

    #pragma unroll
    for (int b = 0; b < 16; ++b) {
      const int base = (b * 256 + c) * 6;
      v2h p0 = *(const v2h*)&pre[base];
      v2h p1 = *(const v2h*)&pre[base + 2];
      v2h p2 = *(const v2h*)&pre[base + 4];
      float r = sigm((float)p0[1] + bir + (float)p0[0] + bhr);
      float z = sigm((float)p1[1] + biz + (float)p1[0] + bhz);
      float n = tanh_fast((float)p2[1] + bin + r * ((float)p2[0] + bhn));
      h[b] = (1.f - z) * n + z * h[b];
      af8[abase + b * 64] = pack_fp8(h[b]);
    }
    __syncthreads();
  }

  #pragma unroll
  for (int b = 0; b < 16; ++b)
    h_last[(wg * 16 + b) * 256 + c] = h[b];
}

__global__ void fc_out(const float* __restrict__ h_last,
                       const float* __restrict__ w_fc,
                       const float* __restrict__ b_fc,
                       float* __restrict__ out) {
  int b = blockIdx.x * blockDim.x + threadIdx.x;
  if (b < 512) {
    float s = 0.f;
    #pragma unroll 8
    for (int k = 0; k < 256; ++k) s += h_last[b * 256 + k] * w_fc[k];
    out[b] = s + b_fc[0];
  }
}

extern "C" void kernel_launch(void* const* d_in, const int* in_sizes, int n_in,
                              void* d_out, int out_size, void* d_ws, size_t ws_size,
                              hipStream_t stream) {
  const float* x      = (const float*)d_in[0];
  const float* w_ih0  = (const float*)d_in[1];
  const float* w_hh0  = (const float*)d_in[2];
  const float* b_ih0  = (const float*)d_in[3];
  const float* b_hh0  = (const float*)d_in[4];
  const float* w_ih1  = (const float*)d_in[5];
  const float* w_hh1  = (const float*)d_in[6];
  const float* b_ih1  = (const float*)d_in[7];
  const float* b_hh1  = (const float*)d_in[8];
  const float* w_fc   = (const float*)d_in[9];
  const float* b_fc   = (const float*)d_in[10];

  unsigned char* ws = (unsigned char*)d_ws;
  v4i*   whh0p = (v4i*)(ws + 0);
  v4i*   whh1p = (v4i*)(ws + 196608);
  v4i*   wih1p = (v4i*)(ws + 393216);
  float* hlast = (float*)(ws + 655360);
  v4i*   h0s   = (v4i*)(ws + (size_t)(2u << 20));

  pack_wb<<<576, 256, 0, stream>>>(w_hh0, w_hh1, w_ih1, (unsigned*)ws);
  gru_layer0<<<NWG, 256, 0, stream>>>(x, w_ih0, b_ih0, b_hh0, whh0p, h0s);
  gru_layer1<<<NWG, 256, 0, stream>>>(b_ih1, b_hh1, whh1p, wih1p, h0s, hlast);
  fc_out<<<2, 256, 0, stream>>>(hlast, w_fc, b_fc, (float*)d_out);
}
